// RelationExtractionModel_73839077753077
// MI455X (gfx1250) — compile-verified, matrix operand order checked by reading, not run on hardware
//
#include <hip/hip_runtime.h>
#include <hip/hip_bf16.h>
#include <math.h>

// ---------------------------------------------------------------------------
// Model dims
// ---------------------------------------------------------------------------
#define BB_   64
#define LL_   512
#define PB_   768
#define HH_   256
#define FEAT_ 912           // 768+32+32+32+32+16
#define FEATP 928           // padded to multiple of 32 for WMMA K loop
#define TIN_  544           // 2H + 32 (multiple of 32)
#define TOK_  (BB_*LL_)     // 32768 rows

typedef _Float16 half8 __attribute__((ext_vector_type(8)));
typedef _Float16 v16h  __attribute__((ext_vector_type(16)));
typedef float    v8f   __attribute__((ext_vector_type(8)));

__device__ __forceinline__ float sigm(float x) { return 1.0f / (1.0f + __expf(-x)); }

// Load one 16x32 f16 WMMA fragment (A or B^T) for this lane.
// Layout (CDNA5 ISA 7.12.2): lane<16 -> row r, K = {k0..k0+7, k0+16..k0+23};
// lane>=16 -> row r, K = {k0+8..k0+15, k0+24..k0+31}.  `p` must already point
// at row_base + k0 + (lane>=16 ? 8 : 0).
__device__ __forceinline__ v16h ld_frag(const _Float16* p) {
  half8 lo = *reinterpret_cast<const half8*>(p);
  half8 hi = *reinterpret_cast<const half8*>(p + 16);
  v16h r;
#pragma unroll
  for (int i = 0; i < 8; ++i) { r[i] = lo[i]; r[i + 8] = hi[i]; }
  return r;
}

#define WMMA32(a, b, c) \
  __builtin_amdgcn_wmma_f32_16x16x32_f16(false, (a), false, (b), (short)0, (c), false, false)

// ---------------------------------------------------------------------------
// Weight convert + transpose:  src (K x N) f32  ->  dst (N x Kp) f16, K..Kp-1 = 0
// ---------------------------------------------------------------------------
__global__ void k_convt(const float* __restrict__ src, _Float16* __restrict__ dst,
                        int K, int N, int Kp) {
  size_t i = (size_t)blockIdx.x * blockDim.x + threadIdx.x;
  size_t total = (size_t)N * Kp;
  if (i >= total) return;
  int n = (int)(i / Kp), k = (int)(i % Kp);
  dst[i] = (k < K) ? (_Float16)src[(size_t)k * N + n] : (_Float16)0.0f;
}

// ---------------------------------------------------------------------------
// Build combined feature rows (TOK_ x FEATP), f16, pad zeroed
// ---------------------------------------------------------------------------
__global__ void k_build(const float* __restrict__ phob, const int* __restrict__ pos_ids,
                        const int* __restrict__ dep_ids, const int* __restrict__ e1,
                        const int* __restrict__ e2, const int* __restrict__ sdp,
                        const float* __restrict__ pos_tab, const float* __restrict__ dep_tab,
                        const float* __restrict__ dist_tab, const float* __restrict__ sdp_tab,
                        _Float16* __restrict__ comb) {
  int t = blockIdx.x;                 // token id 0..TOK_-1
  int b = t / LL_, l = t % LL_;
  int d1 = l - e1[b] + 10; d1 = d1 < 0 ? 0 : (d1 > 20 ? 20 : d1);
  int d2 = l - e2[b] + 10; d2 = d2 < 0 ? 0 : (d2 > 20 ? 20 : d2);
  int pid = pos_ids[t], did = dep_ids[t], sid = sdp[t];
  _Float16* out = comb + (size_t)t * FEATP;
  const float* tok = phob + ((size_t)b * (LL_ + 2) + (l + 1)) * PB_;
  for (int c = threadIdx.x; c < FEATP; c += blockDim.x) {
    float v;
    if      (c < 768) v = tok[c];
    else if (c < 800) v = pos_tab[pid * 32 + (c - 768)];
    else if (c < 832) v = dep_tab[did * 32 + (c - 800)];
    else if (c < 864) v = dist_tab[d1 * 32 + (c - 832)];
    else if (c < 896) v = dist_tab[d2 * 32 + (c - 864)];
    else if (c < 912) v = sdp_tab[sid * 16 + (c - 896)];
    else              v = 0.0f;
    out[c] = (_Float16)v;
  }
}

// Fill dep embedding into tree_x cols [512,544)
__global__ void k_depfill(const int* __restrict__ dep_ids, const float* __restrict__ dep_tab,
                          _Float16* __restrict__ tree_x) {
  size_t i = (size_t)blockIdx.x * blockDim.x + threadIdx.x;
  if (i >= (size_t)TOK_ * 32) return;
  int t = (int)(i >> 5), j = (int)(i & 31);
  tree_x[(size_t)t * TIN_ + 512 + j] = (_Float16)dep_tab[dep_ids[t] * 32 + j];
}

// ---------------------------------------------------------------------------
// Generic WMMA GEMM: C(MxN) = A(MxK,f16,row-major) * Bt(NxK,f16)^T + bias
// One 16x64 strip (4 N-tiles) per wave: A fragment loaded once per k-step and
// reused across 4 WMMAs. 8 waves per block. Requires N % 64 == 0, K % 32 == 0.
// ---------------------------------------------------------------------------
__global__ void k_wmma_gemm(const _Float16* __restrict__ A, int lda,
                            const _Float16* __restrict__ Bt, int ldb,
                            const float* __restrict__ bias,
                            float* __restrict__ C, int ldc,
                            int M, int N, int K) {
  int wid = blockIdx.x * (blockDim.x >> 5) + (threadIdx.x >> 5);
  int nstrips = N >> 6;
  int total = (M >> 4) * nstrips;
  if (wid >= total) return;                     // uniform per wave
  int tm = wid / nstrips, ts = wid % nstrips;
  int lane = threadIdx.x & 31;
  int hs = lane >> 4, r = lane & 15;
  const _Float16* Ar = A + (size_t)(tm * 16 + r) * lda + hs * 8;
  const _Float16* B0 = Bt + (size_t)(ts * 64 + r) * ldb + hs * 8;
  v8f acc0 = {}, acc1 = {}, acc2 = {}, acc3 = {};
  for (int k0 = 0; k0 < K; k0 += 32) {
    v16h a  = ld_frag(Ar + k0);
    v16h b0 = ld_frag(B0 + k0);
    v16h b1 = ld_frag(B0 + (size_t)16 * ldb + k0);
    v16h b2 = ld_frag(B0 + (size_t)32 * ldb + k0);
    v16h b3 = ld_frag(B0 + (size_t)48 * ldb + k0);
    acc0 = WMMA32(a, b0, acc0);
    acc1 = WMMA32(a, b1, acc1);
    acc2 = WMMA32(a, b2, acc2);
    acc3 = WMMA32(a, b3, acc3);
  }
  v8f accs[4] = {acc0, acc1, acc2, acc3};
#pragma unroll
  for (int q = 0; q < 4; ++q) {
    int col = ts * 64 + q * 16 + r;
    float bv = bias ? bias[col] : 0.0f;
#pragma unroll
    for (int v = 0; v < 8; ++v) {
      int row = tm * 16 + hs * 8 + v;
      C[(size_t)row * ldc + col] = accs[q][v] + bv;
    }
  }
}

// ---------------------------------------------------------------------------
// LayerNorm + activation. mode 0 = tanh, 1 = relu. out16/out32 nullable.
// one block (256 thr) per row
// ---------------------------------------------------------------------------
__global__ void k_ln_act(const float* __restrict__ in, const float* __restrict__ gamma,
                         const float* __restrict__ beta, _Float16* __restrict__ out16,
                         float* __restrict__ out32, int N, int mode) {
  __shared__ float r1[256], r2[256];
  int row = blockIdx.x, tid = threadIdx.x;
  const float* x = in + (size_t)row * N;
  float s = 0.0f, s2 = 0.0f;
  for (int i = tid; i < N; i += 256) { float v = x[i]; s += v; s2 += v * v; }
  r1[tid] = s; r2[tid] = s2; __syncthreads();
  for (int st = 128; st > 0; st >>= 1) {
    if (tid < st) { r1[tid] += r1[tid + st]; r2[tid] += r2[tid + st]; }
    __syncthreads();
  }
  float mean = r1[0] / N;
  float var = r2[0] / N - mean * mean;
  float inv = rsqrtf(var + 1e-5f);
  for (int i = tid; i < N; i += 256) {
    float y = (x[i] - mean) * inv * gamma[i] + beta[i];
    y = mode ? fmaxf(y, 0.0f) : tanhf(y);
    if (out16) out16[(size_t)row * N + i] = (_Float16)y;
    if (out32) out32[(size_t)row * N + i] = y;
  }
}

// ---------------------------------------------------------------------------
// BiLSTM recurrence. 4 blocks x 16 batches, 256 threads (8 waves).
// pre: (B,L,1024) f32.  WhT: (1024 x 256) f16 (N-major).  h -> tree_x[col0..]
// A fragments (h state, 8 k-steps) hoisted to registers once per step.
// ---------------------------------------------------------------------------
__global__ void k_lstm(const float* __restrict__ pre, const _Float16* __restrict__ WhT,
                       _Float16* __restrict__ tree_x, int col0, int reverse) {
  __shared__ __align__(16) _Float16 h16[16 * HH_];     // 8 KB
  __shared__ __align__(16) _Float16 g16[16 * 1024];    // 32 KB
  int tid = threadIdx.x;
  int lane = tid & 31, hs = lane >> 4, r = lane & 15;
  int wave = tid >> 5;
  int bb = tid >> 4, u0 = (tid & 15) * 16;             // gate-math mapping
  int bg = blockIdx.x * 16 + bb;                       // global batch
  const _Float16* Ar = h16 + (size_t)r * HH_ + hs * 8;
  float creg[16];
#pragma unroll
  for (int j = 0; j < 16; ++j) creg[j] = 0.0f;
  for (int i = tid; i < 16 * HH_; i += 256) h16[i] = (_Float16)0.0f;
  __syncthreads();

  for (int s = 0; s < LL_; ++s) {
    int t = reverse ? (LL_ - 1 - s) : s;
    // ---- hoist A fragments (h state) for all 8 k-steps ----
    v16h afr[8];
#pragma unroll
    for (int k = 0; k < 8; ++k) afr[k] = ld_frag(Ar + k * 32);
    // ---- g = h @ Wh  (16 x 1024 x 256), 8 N-tiles per wave ----
#pragma unroll
    for (int j = 0; j < 8; ++j) {
      int c0 = (wave * 8 + j) * 16;
      const _Float16* Br = WhT + (size_t)(c0 + r) * HH_ + hs * 8;
      v8f acc = {};
#pragma unroll
      for (int k = 0; k < 8; ++k) {
        v16h b = ld_frag(Br + k * 32);
        acc = WMMA32(afr[k], b, acc);
      }
#pragma unroll
      for (int v = 0; v < 8; ++v)
        g16[(hs * 8 + v) * 1024 + c0 + r] = (_Float16)acc[v];
    }
    __syncthreads();
    // ---- gate math ----
    size_t base = ((size_t)bg * LL_ + t) * 1024;
#pragma unroll
    for (int j = 0; j < 16; ++j) {
      int u = u0 + j;
      float ig = (float)g16[bb * 1024 + u]       + pre[base + u];
      float fg = (float)g16[bb * 1024 + 256 + u] + pre[base + 256 + u];
      float gg = (float)g16[bb * 1024 + 512 + u] + pre[base + 512 + u];
      float og = (float)g16[bb * 1024 + 768 + u] + pre[base + 768 + u];
      float c = sigm(fg) * creg[j] + sigm(ig) * tanhf(gg);
      float h = sigm(og) * tanhf(c);
      creg[j] = c;
      h16[bb * HH_ + u] = (_Float16)h;
      tree_x[((size_t)bg * LL_ + t) * TIN_ + col0 + u] = (_Float16)h;
    }
    __syncthreads();
  }
}

// ---------------------------------------------------------------------------
// Child-sum tree LSTM scan. 4 blocks x 16 batches, 256 threads.
// ---------------------------------------------------------------------------
__global__ void k_tree(const float* __restrict__ xiou, const float* __restrict__ xf,
                       const int* __restrict__ heads,
                       const _Float16* __restrict__ UiouT, const _Float16* __restrict__ UfgT,
                       float* __restrict__ hsum, float* __restrict__ fcsum,
                       float* __restrict__ root) {
  __shared__ __align__(16) _Float16 a16[16 * HH_];     // hsum tile / f-pre reuse (8 KB)
  __shared__ __align__(16) _Float16 g16[16 * 768];     // 24 KB
  __shared__ __align__(16) _Float16 h16[16 * HH_];     // 8 KB
  int tid = threadIdx.x;
  int lane = tid & 31, hs = lane >> 4, r = lane & 15;
  int wave = tid >> 5;
  int bb = tid >> 4, u0 = (tid & 15) * 16;
  int bg = blockIdx.x * 16 + bb;
  const _Float16* Aa = a16 + (size_t)r * HH_ + hs * 8;
  const _Float16* Ah = h16 + (size_t)r * HH_ + hs * 8;

  // zero this block's hsum/fcsum region
  size_t zbase = (size_t)blockIdx.x * 16 * LL_ * HH_;
  for (size_t i = tid; i < (size_t)16 * LL_ * HH_; i += 256) {
    hsum[zbase + i] = 0.0f; fcsum[zbase + i] = 0.0f;
  }
  __threadfence_block();
  __syncthreads();

  float creg[16], hreg[16];
  for (int i = 0; i < LL_; ++i) {
    // load hsum[:, i, :] tile -> a16
#pragma unroll
    for (int j = 0; j < 16; ++j)
      a16[bb * HH_ + u0 + j] = (_Float16)hsum[((size_t)bg * LL_ + i) * HH_ + u0 + j];
    __syncthreads();
    // ---- g = hsum_i @ U_iou  (16 x 768 x 256), 6 N-tiles per wave ----
    {
      v16h afr[8];
#pragma unroll
      for (int k = 0; k < 8; ++k) afr[k] = ld_frag(Aa + k * 32);
#pragma unroll
      for (int j = 0; j < 6; ++j) {
        int c0 = (wave * 6 + j) * 16;
        const _Float16* Br = UiouT + (size_t)(c0 + r) * HH_ + hs * 8;
        v8f acc = {};
#pragma unroll
        for (int k = 0; k < 8; ++k) {
          v16h b = ld_frag(Br + k * 32);
          acc = WMMA32(afr[k], b, acc);
        }
#pragma unroll
        for (int v = 0; v < 8; ++v)
          g16[(hs * 8 + v) * 768 + c0 + r] = (_Float16)acc[v];
      }
    }
    __syncthreads();
    // ---- gate math 1 ----
    size_t row = (size_t)bg * LL_ + i;
#pragma unroll
    for (int j = 0; j < 16; ++j) {
      int u = u0 + j;
      float ig = (float)g16[bb * 768 + u]       + xiou[row * 768 + u];
      float og = (float)g16[bb * 768 + 256 + u] + xiou[row * 768 + 256 + u];
      float ug = (float)g16[bb * 768 + 512 + u] + xiou[row * 768 + 512 + u];
      float c = sigm(ig) * tanhf(ug) + fcsum[row * HH_ + u];
      float h = sigm(og) * tanhf(c);
      creg[j] = c; hreg[j] = h;
      h16[bb * HH_ + u] = (_Float16)h;
    }
    __syncthreads();
    // ---- fpre = h @ U_fg (16 x 256 x 256), 2 N-tiles per wave, into a16 ----
    {
      v16h afr[8];
#pragma unroll
      for (int k = 0; k < 8; ++k) afr[k] = ld_frag(Ah + k * 32);
#pragma unroll
      for (int j = 0; j < 2; ++j) {
        int c0 = (wave * 2 + j) * 16;
        const _Float16* Br = UfgT + (size_t)(c0 + r) * HH_ + hs * 8;
        v8f acc = {};
#pragma unroll
        for (int k = 0; k < 8; ++k) {
          v16h b = ld_frag(Br + k * 32);
          acc = WMMA32(afr[k], b, acc);
        }
#pragma unroll
        for (int v = 0; v < 8; ++v)
          a16[(hs * 8 + v) * HH_ + c0 + r] = (_Float16)acc[v];
      }
    }
    __syncthreads();
    // ---- gate math 2: scatter to parent ----
    int p = heads[bg * LL_ + i];
#pragma unroll
    for (int j = 0; j < 16; ++j) {
      int u = u0 + j;
      float f = sigm(xf[((size_t)bg * LL_ + p) * HH_ + u] + (float)a16[bb * HH_ + u]);
      if (i < LL_ - 1) {
        hsum[((size_t)bg * LL_ + p) * HH_ + u]  += hreg[j];
        fcsum[((size_t)bg * LL_ + p) * HH_ + u] += f * creg[j];
      }
      if (i == LL_ - 1) root[(size_t)bg * HH_ + u] = hreg[j];
    }
    __threadfence_block();
    __syncthreads();
  }
}

// ---------------------------------------------------------------------------
// Small scalar f32 GEMM for the 64-row tail MLP
// ---------------------------------------------------------------------------
__global__ void k_sgemm(const float* __restrict__ A, const float* __restrict__ W,
                        const float* __restrict__ bias, float* __restrict__ C,
                        int M, int N, int K) {
  int i = blockIdx.x * blockDim.x + threadIdx.x;
  if (i >= M * N) return;
  int m = i / N, n = i % N;
  float acc = bias[n];
  for (int k = 0; k < K; ++k) acc += A[(size_t)m * K + k] * W[(size_t)k * N + n];
  C[i] = acc;
}

// ---------------------------------------------------------------------------
// Launcher
// ---------------------------------------------------------------------------
extern "C" void kernel_launch(void* const* d_in, const int* in_sizes, int n_in,
                              void* d_out, int out_size, void* d_ws, size_t ws_size,
                              hipStream_t stream) {
  (void)in_sizes; (void)n_in; (void)out_size; (void)ws_size;
  const float* phob     = (const float*)d_in[0];
  const int*   pos_ids  = (const int*)d_in[1];
  const int*   dep_ids  = (const int*)d_in[2];
  const int*   e1       = (const int*)d_in[3];
  const int*   e2       = (const int*)d_in[4];
  const int*   sdp      = (const int*)d_in[5];
  const int*   heads    = (const int*)d_in[6];
  const float* pos_tab  = (const float*)d_in[7];
  const float* dep_tab  = (const float*)d_in[8];
  const float* dist_tab = (const float*)d_in[9];
  const float* sdp_tab  = (const float*)d_in[10];
  const float* ft_W     = (const float*)d_in[11];
  const float* ft_b     = (const float*)d_in[12];
  const float* ln1_g    = (const float*)d_in[13];
  const float* ln1_b    = (const float*)d_in[14];
  const float* Wx_f     = (const float*)d_in[15];
  const float* Wh_f     = (const float*)d_in[16];
  const float* bl_f     = (const float*)d_in[17];
  const float* Wx_b     = (const float*)d_in[18];
  const float* Wh_b     = (const float*)d_in[19];
  const float* bl_b     = (const float*)d_in[20];
  const float* W_iou    = (const float*)d_in[21];
  const float* U_iou    = (const float*)d_in[22];
  const float* b_iou    = (const float*)d_in[23];
  const float* W_fg     = (const float*)d_in[24];
  const float* U_fg     = (const float*)d_in[25];
  const float* b_fg     = (const float*)d_in[26];
  const float* W1       = (const float*)d_in[27];
  const float* b1       = (const float*)d_in[28];
  const float* ln2_g    = (const float*)d_in[29];
  const float* ln2_b    = (const float*)d_in[30];
  const float* W2       = (const float*)d_in[31];
  const float* b2       = (const float*)d_in[32];
  const float* ln3_g    = (const float*)d_in[33];
  const float* ln3_b    = (const float*)d_in[34];
  const float* W3       = (const float*)d_in[35];
  const float* b3       = (const float*)d_in[36];
  float* out = (float*)d_out;

  // workspace layout
  size_t off = 0;
  char* base = (char*)d_ws;
  auto alloc = [&](size_t bytes) -> char* {
    char* p = base + off; off += (bytes + 255) & ~(size_t)255; return p;
  };
  _Float16* wt_ftW  = (_Float16*)alloc((size_t)512 * FEATP * 2);
  _Float16* wt_WxF  = (_Float16*)alloc((size_t)1024 * 512 * 2);
  _Float16* wt_WxB  = (_Float16*)alloc((size_t)1024 * 512 * 2);
  _Float16* wt_WhF  = (_Float16*)alloc((size_t)1024 * 256 * 2);
  _Float16* wt_WhB  = (_Float16*)alloc((size_t)1024 * 256 * 2);
  _Float16* wt_Wiou = (_Float16*)alloc((size_t)768 * TIN_ * 2);
  _Float16* wt_Uiou = (_Float16*)alloc((size_t)768 * 256 * 2);
  _Float16* wt_Wfg  = (_Float16*)alloc((size_t)256 * TIN_ * 2);
  _Float16* wt_Ufg  = (_Float16*)alloc((size_t)256 * 256 * 2);
  _Float16* comb    = (_Float16*)alloc((size_t)TOK_ * FEATP * 2);
  float*    xpre    = (float*)   alloc((size_t)TOK_ * 512 * 4);
  _Float16* x16     = (_Float16*)alloc((size_t)TOK_ * 512 * 2);
  float*    preF    = (float*)   alloc((size_t)TOK_ * 1024 * 4);
  float*    preB    = (float*)   alloc((size_t)TOK_ * 1024 * 4);
  _Float16* tree_x  = (_Float16*)alloc((size_t)TOK_ * TIN_ * 2);
  float*    xiou    = (float*)   alloc((size_t)TOK_ * 768 * 4);
  float*    xf      = (float*)   alloc((size_t)TOK_ * 256 * 4);
  float*    hsum    = (float*)   alloc((size_t)TOK_ * 256 * 4);
  float*    fcsum   = (float*)   alloc((size_t)TOK_ * 256 * 4);
  float*    root    = (float*)   alloc((size_t)BB_ * 256 * 4);
  float*    t1      = (float*)   alloc((size_t)BB_ * 512 * 4);
  float*    h1      = (float*)   alloc((size_t)BB_ * 512 * 4);
  float*    t2      = (float*)   alloc((size_t)BB_ * 256 * 4);
  float*    h2      = (float*)   alloc((size_t)BB_ * 256 * 4);

  const int TB = 256;
  auto grid1 = [&](size_t total) { return (unsigned)((total + TB - 1) / TB); };

  // 1) weight conversion (K x N f32 -> N x Kp f16)
  k_convt<<<grid1((size_t)512 * FEATP), TB, 0, stream>>>(ft_W,  wt_ftW,  FEAT_, 512, FEATP);
  k_convt<<<grid1((size_t)1024 * 512), TB, 0, stream>>>(Wx_f,  wt_WxF,  512, 1024, 512);
  k_convt<<<grid1((size_t)1024 * 512), TB, 0, stream>>>(Wx_b,  wt_WxB,  512, 1024, 512);
  k_convt<<<grid1((size_t)1024 * 256), TB, 0, stream>>>(Wh_f,  wt_WhF,  256, 1024, 256);
  k_convt<<<grid1((size_t)1024 * 256), TB, 0, stream>>>(Wh_b,  wt_WhB,  256, 1024, 256);
  k_convt<<<grid1((size_t)768 * TIN_), TB, 0, stream>>>(W_iou, wt_Wiou, TIN_, 768, TIN_);
  k_convt<<<grid1((size_t)768 * 256),  TB, 0, stream>>>(U_iou, wt_Uiou, 256, 768, 256);
  k_convt<<<grid1((size_t)256 * TIN_), TB, 0, stream>>>(W_fg,  wt_Wfg,  TIN_, 256, TIN_);
  k_convt<<<grid1((size_t)256 * 256),  TB, 0, stream>>>(U_fg,  wt_Ufg,  256, 256, 256);

  // 2) combined features
  k_build<<<TOK_, 128, 0, stream>>>(phob, pos_ids, dep_ids, e1, e2, sdp,
                                    pos_tab, dep_tab, dist_tab, sdp_tab, comb);

  // 3) xpre = comb @ ft_W + ft_b   (32768 x 512, K=928)
  {
    unsigned waves = (TOK_ / 16) * (512 / 64);
    k_wmma_gemm<<<waves / 8, TB, 0, stream>>>(comb, FEATP, wt_ftW, FEATP, ft_b,
                                              xpre, 512, TOK_, 512, FEATP);
  }
  // 4) x = tanh(LN(xpre))  -> f16
  k_ln_act<<<TOK_, TB, 0, stream>>>(xpre, ln1_g, ln1_b, x16, nullptr, 512, 0);

  // 5) LSTM input projections
  {
    unsigned waves = (TOK_ / 16) * (1024 / 64);
    k_wmma_gemm<<<waves / 8, TB, 0, stream>>>(x16, 512, wt_WxF, 512, bl_f,
                                              preF, 1024, TOK_, 1024, 512);
    k_wmma_gemm<<<waves / 8, TB, 0, stream>>>(x16, 512, wt_WxB, 512, bl_b,
                                              preB, 1024, TOK_, 1024, 512);
  }

  // 6) dep embedding columns of tree_x
  k_depfill<<<grid1((size_t)TOK_ * 32), TB, 0, stream>>>(dep_ids, dep_tab, tree_x);

  // 7) BiLSTM recurrences (sequential scans, batch-tiled WMMA)
  k_lstm<<<4, TB, 0, stream>>>(preF, wt_WhF, tree_x, 0,   0);
  k_lstm<<<4, TB, 0, stream>>>(preB, wt_WhB, tree_x, 256, 1);

  // 8) tree projections
  {
    unsigned waves = (TOK_ / 16) * (768 / 64);
    k_wmma_gemm<<<waves / 8, TB, 0, stream>>>(tree_x, TIN_, wt_Wiou, TIN_, b_iou,
                                              xiou, 768, TOK_, 768, TIN_);
    waves = (TOK_ / 16) * (256 / 64);
    k_wmma_gemm<<<waves / 8, TB, 0, stream>>>(tree_x, TIN_, wt_Wfg, TIN_, b_fg,
                                              xf, 256, TOK_, 256, TIN_);
  }

  // 9) tree LSTM scan
  k_tree<<<4, TB, 0, stream>>>(xiou, xf, heads, wt_Uiou, wt_Ufg, hsum, fcsum, root);

  // 10) tail MLP (tiny)
  k_sgemm<<<grid1((size_t)BB_ * 512), TB, 0, stream>>>(root, W1, b1, t1, BB_, 512, 256);
  k_ln_act<<<BB_, TB, 0, stream>>>(t1, ln2_g, ln2_b, nullptr, h1, 512, 1);
  k_sgemm<<<grid1((size_t)BB_ * 256), TB, 0, stream>>>(h1, W2, b2, t2, BB_, 256, 512);
  k_ln_act<<<BB_, TB, 0, stream>>>(t2, ln3_g, ln3_b, nullptr, h2, 256, 1);
  k_sgemm<<<grid1((size_t)BB_ * 10), TB, 0, stream>>>(h2, W3, b3, out, BB_, 10, 256);
}
